// ResiduesDecode_16363825397924
// MI455X (gfx1250) — matrix operand order repaired and checked by reading, not compiled, not run on hardware
//
#include <hip/hip_runtime.h>

// ---------------- problem constants ----------------
#define B_BATCH 64
#define R_RES   512
#define VDIM    256
#define NLET    20
#define MAXA    13            // MAX_NATOM
#define NCOLS   48            // (b*3+c) columns handled per pass
#define NPASS   4             // 4 * 48 = 192 = B_BATCH*3
#define NT_W    3             // 16-wide N tiles per pass (48/16)
#define NTHREADS 512          // 16 wave32 waves

typedef _Float16 half4  __attribute__((ext_vector_type(4)));
typedef _Float16 half8  __attribute__((ext_vector_type(8)));
typedef _Float16 half16 __attribute__((ext_vector_type(16)));
typedef float    floatx4 __attribute__((ext_vector_type(4)));
typedef float    v8f     __attribute__((ext_vector_type(8)));

__constant__ int NATOMS_D[NLET] = {4,10,7,7,5,8,8,3,9,7,7,8,7,10,6,5,6,13,11,6};

// -------- fragment helpers --------
__device__ __forceinline__ half8 cvt8(const float* p) {
  const floatx4* q = (const floatx4*)p;
  floatx4 a = q[0], b = q[1];
  half8 r;
  r[0]=(_Float16)a[0]; r[1]=(_Float16)a[1]; r[2]=(_Float16)a[2]; r[3]=(_Float16)a[3];
  r[4]=(_Float16)b[0]; r[5]=(_Float16)b[1]; r[6]=(_Float16)b[2]; r[7]=(_Float16)b[3];
  return r;
}

__device__ __forceinline__ half16 cat8(half8 a, half8 b) {
  half16 r;
#pragma unroll
  for (int i = 0; i < 8; ++i) { r[i] = a[i]; r[i + 8] = b[i]; }
  return r;
}

// A (16x32 f16, row-major source): lanes 0-15 hold K [kb..kb+7] + [kb+16..kb+23],
// lanes 16-31 hold K [kb+8..kb+15] + [kb+24..kb+31]  (ISA 16-bit A layout)
__device__ __forceinline__ half16 afrag_from_f32(const float* row, int kb, int hsel) {
  int h8 = hsel >> 1;                       // 0 or 8
  return cat8(cvt8(row + kb + h8), cvt8(row + kb + 16 + h8));
}

// B (32x16): lane n holds column n; low lanes K [kb..kb+7] + [kb+16..kb+23], high lanes K [kb+8..kb+15] + [kb+24..kb+31]
__device__ __forceinline__ half16 bfrag_from_lds(const _Float16* rowp, int kb, int hsel) {
  const _Float16* q = rowp + kb + (hsel >> 1);
  return cat8(*(const half8*)q, *(const half8*)(q + 16));
}

// -------- prefix-sum over residues: abase[r] = sum_{r'<r} natom(seq[r']) --------
extern "C" __global__ __launch_bounds__(R_RES)
void residues_scan(const int* __restrict__ seq, int* __restrict__ abase) {
  __shared__ int s[R_RES];
  int t = threadIdx.x;
  int na = NATOMS_D[seq[t]];
  s[t] = na;
  __syncthreads();
  for (int off = 1; off < R_RES; off <<= 1) {
    int v = (t >= off) ? s[t - off] : 0;
    __syncthreads();
    s[t] += v;
    __syncthreads();
  }
  abase[t] = s[t] - na;                     // exclusive prefix
}

// -------- main kernel: one workgroup per residue --------
extern "C" __global__ __launch_bounds__(NTHREADS)
void residues_decode_wmma(const float* __restrict__ pos_ca,
                          const float* __restrict__ x_v,
                          const int*   __restrict__ seq,
                          const float* __restrict__ W1,
                          const float* __restrict__ W2,
                          const int*   __restrict__ abase,
                          float* __restrict__ out,
                          int total_atoms)
{
  __shared__ _Float16 Xt[NCOLS * VDIM];   // X^T  : [n][k]  (24 KB)
  __shared__ _Float16 Ht[NCOLS * VDIM];   // H^T  : [n][m]  (24 KB)

  const int r     = blockIdx.x;
  const int l     = seq[r];
  const int natom = NATOMS_D[l];
  const int ab    = abase[r];
  const int tid   = threadIdx.x;
  const int w     = tid >> 5;             // wave id, 0..15
  const int lane  = tid & 31;
  const int lr    = lane & 15;
  const int hsel  = lane & 16;            // 0 / 16
  const int m0    = hsel >> 1;            // 0 / 8 : C-tile row base for this lane half

  const float* A1row = W1 + ((size_t)l << 16) + (size_t)(w * 16 + lr) * VDIM;
  const float* A2row = W2 + ((size_t)l * MAXA + lr) * VDIM;

  for (int pass = 0; pass < NPASS; ++pass) {
    __syncthreads();                      // LDS safe to overwrite

    // ---- stage Xt[n][k] = f16(x_v[b, r, k, c]),  n = b*3+c (48 cols this pass) ----
    for (int idx = tid; idx < NCOLS * (VDIM / 4); idx += NTHREADS) {
      int nn = idx >> 6;                  // 0..47
      int k4 = (idx & 63) << 2;           // 0..252
      int n  = pass * NCOLS + nn;
      int b  = n / 3;
      int c  = n - b * 3;
      const float* src = x_v + ((size_t)b * R_RES + r) * (VDIM * 3) + c;
      half4 hv;
#pragma unroll
      for (int i = 0; i < 4; ++i) hv[i] = (_Float16)src[(size_t)(k4 + i) * 3];
      *(half4*)&Xt[nn * VDIM + k4] = hv;
    }
    __syncthreads();

    // ---- GEMM1: H = W1[l] (256x256) x X (256x48); wave w owns M rows [16w,16w+16) ----
    v8f acc[NT_W];
#pragma unroll
    for (int nt = 0; nt < NT_W; ++nt) acc[nt] = {};
    for (int kt = 0; kt < VDIM / 32; ++kt) {
      int kb = kt * 32;
      __builtin_prefetch(A1row + kb + 32, 0, 1);    // global_prefetch_b8: next K slab of W1
      half16 af = afrag_from_f32(A1row, kb, hsel);
#pragma unroll
      for (int nt = 0; nt < NT_W; ++nt) {
        half16 bf = bfrag_from_lds(&Xt[(nt * 16 + lr) * VDIM], kb, hsel);
        acc[nt] = __builtin_amdgcn_wmma_f32_16x16x32_f16(
            false, af, false, bf, (short)0, acc[nt], false, false);
      }
    }
    // write H^T to LDS (f32 -> f16), contiguous 8 M-values per lane
#pragma unroll
    for (int nt = 0; nt < NT_W; ++nt) {
      half8 hv;
#pragma unroll
      for (int j = 0; j < 8; ++j) hv[j] = (_Float16)acc[nt][j];
      *(half8*)&Ht[(nt * 16 + lr) * VDIM + w * 16 + m0] = hv;
    }
    __syncthreads();

    // ---- GEMM2 (waves 0..2): Out = W2[l] (16-pad x 256) x H (256x48), fused scatter ----
    if (w < NT_W) {
      v8f acc2 = {};
      for (int kt = 0; kt < VDIM / 32; ++kt) {
        int kb = kt * 32;
        half16 a2 = {};
        if (lr < MAXA) a2 = afrag_from_f32(A2row, kb, hsel);   // rows >= 13 stay zero
        half16 b2 = bfrag_from_lds(&Ht[(w * 16 + lr) * VDIM], kb, hsel);
        acc2 = __builtin_amdgcn_wmma_f32_16x16x32_f16(
            false, a2, false, b2, (short)0, acc2, false, false);
      }
      // column n -> (b, c); rows m0+j -> atom index; add CA, scatter ragged
      int n = pass * NCOLS + w * 16 + lr;
      int b = n / 3;
      int c = n - b * 3;
      float pca = pos_ca[((size_t)b * R_RES + r) * 3 + c];
#pragma unroll
      for (int j = 0; j < 8; ++j) {
        int a = m0 + j;
        if (a < natom)
          out[((size_t)b * total_atoms + ab + a) * 3 + c] = acc2[j] + pca;
      }
    }
  }
}

extern "C" void kernel_launch(void* const* d_in, const int* in_sizes, int n_in,
                              void* d_out, int out_size, void* d_ws, size_t ws_size,
                              hipStream_t stream) {
  const float* pos_ca = (const float*)d_in[0];
  const float* x_v    = (const float*)d_in[1];
  const int*   seq    = (const int*)d_in[2];
  const float* W1     = (const float*)d_in[3];
  const float* W2     = (const float*)d_in[4];
  float* out = (float*)d_out;
  int*   abase = (int*)d_ws;

  const int total_atoms = out_size / (B_BATCH * 3);

  residues_scan<<<1, R_RES, 0, stream>>>(seq, abase);
  residues_decode_wmma<<<R_RES, NTHREADS, 0, stream>>>(
      pos_ca, x_v, seq, W1, W2, abase, out, total_atoms);

  (void)in_sizes; (void)n_in; (void)ws_size;
}
